// PromiseGraphTransformer_56865366999290
// MI455X (gfx1250) — matrix operand order repaired and checked by reading, not compiled, not run on hardware
//
#include <hip/hip_runtime.h>
#include <hip/hip_bf16.h>
#include <math.h>

// ---------------------------------------------------------------------------
// Types for WMMA operands (wave32, gfx1250)
// ---------------------------------------------------------------------------
typedef __attribute__((ext_vector_type(16))) _Float16 v16h;
typedef __attribute__((ext_vector_type(8)))  _Float16 v8h;
typedef __attribute__((ext_vector_type(4)))  _Float16 v4h;
typedef __attribute__((ext_vector_type(8)))  float    v8f;

#define CDIM 64     // per-head dim == embedding dim
#define DDIM 256    // H*C
#define HEADS 4

// order-preserving float->uint encoding for atomic max on signed floats
__device__ __forceinline__ unsigned enc_f32(float f) {
    unsigned u = __float_as_uint(f);
    return (u >> 31) ? ~u : (u | 0x80000000u);
}
__device__ __forceinline__ float dec_f32(unsigned e) {
    return (e >> 31) ? __uint_as_float(e ^ 0x80000000u) : __uint_as_float(~e);
}
#define ENC_NEG_INF 0x007FFFFFu   // enc(-inf)

__device__ __forceinline__ float sigmoidf_(float x) { return 1.0f / (1.0f + __expf(-x)); }

// ---------------------------------------------------------------------------
// Kernel 1: repack the four [64,256] fp32 weight matrices into the wave32
// WMMA-B f16 operand layout.  Layout consumed by qkv_wmma:
//   half index g = tile*1024 + kb*512 + lane*16 + j
//   tile = mat*16 + colTile (64 tiles), kb = K-block (0: K 0..31, 1: K 32..63)
//   lane j half -> W[K][n] with K = kb*32 + (lane>=16 ? 8 : 0) + (j&7) + (j>=8 ? 16 : 0),
//                              n = colTile*16 + (lane & 15)
// (16-bit B 32x16: lanes 0-15 hold K=0..7,16..23, lanes 16-31 hold K=8..15,24..31,
//  two K values packed per VGPR -- ISA 7.12.2)
// ---------------------------------------------------------------------------
__global__ void pack_w_kernel(const float* __restrict__ Wq,
                              const float* __restrict__ Wk,
                              const float* __restrict__ Wv,
                              const float* __restrict__ Wskip,
                              _Float16* __restrict__ wpack) {
    int g = blockIdx.x * blockDim.x + threadIdx.x;   // [0, 65536)
    if (g >= 4 * CDIM * DDIM) return;
    int tile   = g >> 10;
    int within = g & 1023;
    int kb     = within >> 9;
    int rem    = within & 511;
    int lane   = rem >> 4;
    int j      = rem & 15;
    int K = kb * 32 + ((lane & 16) ? 8 : 0) + (j & 7) + ((j & 8) ? 16 : 0);
    int n = (tile & 15) * 16 + (lane & 15);
    int mat = tile >> 4;
    const float* W = (mat == 0) ? Wq : (mat == 1) ? Wk : (mat == 2) ? Wv : Wskip;
    wpack[g] = (_Float16)W[K * DDIM + n];
}

// ---------------------------------------------------------------------------
// Kernel 2: embedding gather + fused q/k/v/x_r projection via WMMA.
// Block = 256 threads (8 waves), handles 16 nodes.  Each wave owns 8 of the
// 64 output tiles (4 matrices x 16 column tiles), 2 WMMA (K=32) per tile.
// ---------------------------------------------------------------------------
__global__ void qkv_wmma_kernel(const int*   __restrict__ x,
                                const float* __restrict__ embed,
                                const _Float16* __restrict__ wpack,
                                const float* __restrict__ bq,
                                const float* __restrict__ bk,
                                const float* __restrict__ bv,
                                const float* __restrict__ bskip,
                                float* __restrict__ q,
                                float* __restrict__ k,
                                float* __restrict__ v,
                                float* __restrict__ xr,
                                int N) {
    __shared__ _Float16 hs[16 * CDIM];      // 16-node f16 feature tile, 2 KB

    const int node0 = blockIdx.x * 16;

    // vectorized embedding gather: thread t owns node (t>>4), features
    // (t&15)*4 .. +3 -> one float4 global load, one 8-byte LDS store
    {
        int nl = threadIdx.x >> 4;
        int c4 = (threadIdx.x & 15) * 4;
        int n  = node0 + nl;
        v4h hv = {};
        if (n < N) {
            int idx = x[n];
            float4 f = *(const float4*)(embed + (size_t)idx * CDIM + c4);
            hv[0] = (_Float16)f.x; hv[1] = (_Float16)f.y;
            hv[2] = (_Float16)f.z; hv[3] = (_Float16)f.w;
        }
        *(v4h*)(hs + nl * CDIM + c4) = hv;
    }
    __syncthreads();

    const int wave = threadIdx.x >> 5;
    const int lane = threadIdx.x & 31;
    const int m    = lane & 15;
    const int hi   = lane >> 4;

    // A operand (16-bit 16x32, ISA 7.12.2): lane holds row m, K-base hi*8;
    // halves 0..7 -> K = kbA..kbA+7 ; halves 8..15 -> K = kbA+16..kbA+23
    const _Float16* rowp = &hs[m * CDIM];
    const int kbA = hi * 8;
    v8h a0lo = *(const v8h*)(rowp + kbA);
    v8h a0hi = *(const v8h*)(rowp + kbA + 16);
    v8h a1lo = *(const v8h*)(rowp + kbA + 32);
    v8h a1hi = *(const v8h*)(rowp + kbA + 48);
    v16h A0, A1;
#pragma unroll
    for (int j = 0; j < 8; ++j) {
        A0[j] = a0lo[j]; A0[8 + j] = a0hi[j];
        A1[j] = a1lo[j]; A1[8 + j] = a1hi[j];
    }

    const bool full = (node0 + 16 <= N);    // wave-uniform fast-path flag

#pragma unroll
    for (int i = 0; i < 8; ++i) {
        const int tt  = wave * 8 + i;       // tile id 0..63
        const int mat = tt >> 4;
        const int ct  = tt & 15;

        v16h B0 = *(const v16h*)(wpack + (size_t)(tt * 2 + 0) * 512 + lane * 16);
        v16h B1 = *(const v16h*)(wpack + (size_t)(tt * 2 + 1) * 512 + lane * 16);

        v8f c = {};
        c = __builtin_amdgcn_wmma_f32_16x16x32_f16(false, A0, false, B0,
                                                   (short)0, c, false, false);
        c = __builtin_amdgcn_wmma_f32_16x16x32_f16(false, A1, false, B1,
                                                   (short)0, c, false, false);

        const float* biasp; float* dstp;
        switch (mat) {
            case 0:  biasp = bq;    dstp = q;  break;
            case 1:  biasp = bk;    dstp = k;  break;
            case 2:  biasp = bv;    dstp = v;  break;
            default: biasp = bskip; dstp = xr; break;
        }
        const int col   = ct * 16 + m;          // N index of this lane
        const float bvl = biasp[col];
        const int rbase = node0 + hi * 8;       // D layout: M = r + hi*8
        if (full) {
#pragma unroll
            for (int r = 0; r < 8; ++r)
                dstp[(size_t)(rbase + r) * DDIM + col] = c[r] + bvl;
        } else {
#pragma unroll
            for (int r = 0; r < 8; ++r) {
                int row = rbase + r;
                if (row < N) dstp[(size_t)row * DDIM + col] = c[r] + bvl;
            }
        }
    }
}

// ---------------------------------------------------------------------------
// Kernel 3: clear aggregation buffer + softmax stats
// ---------------------------------------------------------------------------
__global__ void init_kernel(float* __restrict__ out,
                            float* __restrict__ nodesum,
                            unsigned* __restrict__ nodemax,
                            int n256, int n4) {
    int g = blockIdx.x * blockDim.x + threadIdx.x;
    if (g < n256) out[g] = 0.0f;
    if (g < n4) { nodesum[g] = 0.0f; nodemax[g] = ENC_NEG_INF; }
}

// ---------------------------------------------------------------------------
// Kernel 4: per-(edge,head) attention logit + atomic segment max
// ---------------------------------------------------------------------------
__global__ void edge_logits_kernel(const int* __restrict__ ei,
                                   const float* __restrict__ q,
                                   const float* __restrict__ k,
                                   float* __restrict__ logits,
                                   unsigned* __restrict__ nodemax,
                                   int E) {
    int g = blockIdx.x * blockDim.x + threadIdx.x;
    if (g >= E * HEADS) return;
    int e = g >> 2, h = g & 3;
    int src = ei[e], dst = ei[E + e];
    const float4* qp = (const float4*)(q + (size_t)dst * DDIM + h * CDIM);
    const float4* kp = (const float4*)(k + (size_t)src * DDIM + h * CDIM);
    float acc = 0.0f;
#pragma unroll
    for (int i = 0; i < CDIM / 4; ++i) {
        float4 a = qp[i], b = kp[i];
        acc += a.x * b.x + a.y * b.y + a.z * b.z + a.w * b.w;
    }
    float lg = acc * 0.125f;                 // 1/sqrt(64)
    logits[(size_t)e * HEADS + h] = lg;
    atomicMax(&nodemax[(size_t)dst * HEADS + h], enc_f32(lg));
}

// ---------------------------------------------------------------------------
// Kernel 5: exp(logit - max) + atomic segment sum (alpha stored in-place)
// ---------------------------------------------------------------------------
__global__ void edge_exp_kernel(const int* __restrict__ ei,
                                float* __restrict__ logits,
                                const unsigned* __restrict__ nodemax,
                                float* __restrict__ nodesum,
                                int E) {
    int g = blockIdx.x * blockDim.x + threadIdx.x;
    if (g >= E * HEADS) return;
    int e = g >> 2, h = g & 3;
    int dst = ei[E + e];
    float mx = dec_f32(nodemax[(size_t)dst * HEADS + h]);
    float ex = __expf(logits[(size_t)e * HEADS + h] - mx);
    logits[(size_t)e * HEADS + h] = ex;
    atomicAdd(&nodesum[(size_t)dst * HEADS + h], ex);
}

// ---------------------------------------------------------------------------
// Kernel 6: wave-per-edge weighted scatter of v rows (L2-resident atomics)
// ---------------------------------------------------------------------------
__global__ void edge_agg_kernel(const int* __restrict__ ei,
                                const float* __restrict__ logits,   // exp values
                                const float* __restrict__ nodesum,
                                const float* __restrict__ v,
                                float* __restrict__ out,
                                int E) {
    int e = blockIdx.x * 8 + (threadIdx.x >> 5);
    if (e >= E) return;                      // wave-uniform guard
    int lane = threadIdx.x & 31;
    int src = ei[e], dst = ei[E + e];
    int h = lane >> 3;                       // 8 components per lane, 1 head
    float ex = logits[(size_t)e * HEADS + h];
    float s  = nodesum[(size_t)dst * HEADS + h];
    float w  = ex / (s + 1e-16f);
    const float4* vp = (const float4*)(v + (size_t)src * DDIM + lane * 8);
    float4 a = vp[0], b = vp[1];
    float* op = out + (size_t)dst * DDIM + lane * 8;
    atomicAdd(op + 0, w * a.x); atomicAdd(op + 1, w * a.y);
    atomicAdd(op + 2, w * a.z); atomicAdd(op + 3, w * a.w);
    atomicAdd(op + 4, w * b.x); atomicAdd(op + 5, w * b.y);
    atomicAdd(op + 6, w * b.z); atomicAdd(op + 7, w * b.w);
}

// ---------------------------------------------------------------------------
// Kernel 7: wave-per-node beta gate + ReLU + sigmoid heads
// ---------------------------------------------------------------------------
__global__ void finalize_kernel(const float* __restrict__ out,
                                const float* __restrict__ xr,
                                const float* __restrict__ Wbeta,
                                const float* __restrict__ Wcoop,
                                const float* __restrict__ bcoop,
                                const float* __restrict__ Wanom,
                                const float* __restrict__ banom,
                                float* __restrict__ coopO,
                                float* __restrict__ anomO,
                                float* __restrict__ hO,
                                int N) {
    int node = blockIdx.x * 8 + (threadIdx.x >> 5);
    if (node >= N) return;                 // wave-uniform guard
    int lane = threadIdx.x & 31;
    size_t base = (size_t)node * DDIM + lane * 8;

    const float4* op = (const float4*)(out + base);
    const float4* xp = (const float4*)(xr + base);
    float4 o0 = op[0], o1 = op[1], x0 = xp[0], x1 = xp[1];
    float ov[8] = { o0.x, o0.y, o0.z, o0.w, o1.x, o1.y, o1.z, o1.w };
    float xv[8] = { x0.x, x0.y, x0.z, x0.w, x1.x, x1.y, x1.z, x1.w };

    float p = 0.0f;
#pragma unroll
    for (int j = 0; j < 8; ++j) {
        int i = lane * 8 + j;
        p += ov[j] * Wbeta[i] + xv[j] * Wbeta[DDIM + i]
           + (ov[j] - xv[j]) * Wbeta[2 * DDIM + i];
    }
#pragma unroll
    for (int off = 16; off >= 1; off >>= 1) p += __shfl_xor(p, off, 32);
    float beta = sigmoidf_(p);

    float pc = 0.0f, pa = 0.0f;
#pragma unroll
    for (int j = 0; j < 8; ++j) {
        int i = lane * 8 + j;
        float hv = beta * xv[j] + (1.0f - beta) * ov[j];
        hv = fmaxf(hv, 0.0f);
        hO[base + j] = hv;
        pc += hv * Wcoop[i];
        pa += hv * Wanom[i];
    }
#pragma unroll
    for (int off = 16; off >= 1; off >>= 1) {
        pc += __shfl_xor(pc, off, 32);
        pa += __shfl_xor(pa, off, 32);
    }
    if (lane == 0) {
        coopO[node] = sigmoidf_(pc + bcoop[0]);
        anomO[node] = sigmoidf_(pa + banom[0]);
    }
}

// ---------------------------------------------------------------------------
// Launcher
// ---------------------------------------------------------------------------
extern "C" void kernel_launch(void* const* d_in, const int* in_sizes, int n_in,
                              void* d_out, int out_size, void* d_ws, size_t ws_size,
                              hipStream_t stream) {
    const int*   x     = (const int*)  d_in[0];
    const int*   ei    = (const int*)  d_in[1];
    const float* embed = (const float*)d_in[2];
    const float* Wq    = (const float*)d_in[3];
    const float* bq    = (const float*)d_in[4];
    const float* Wk    = (const float*)d_in[5];
    const float* bk    = (const float*)d_in[6];
    const float* Wv    = (const float*)d_in[7];
    const float* bv    = (const float*)d_in[8];
    const float* Wskip = (const float*)d_in[9];
    const float* bskip = (const float*)d_in[10];
    const float* Wbeta = (const float*)d_in[11];
    const float* Wcoop = (const float*)d_in[12];
    const float* bcoop = (const float*)d_in[13];
    const float* Wanom = (const float*)d_in[14];
    const float* banom = (const float*)d_in[15];

    const int N = in_sizes[0];
    const int E = in_sizes[1] / 2;

    // ---- workspace layout -------------------------------------------------
    char* ws = (char*)d_ws;
    size_t off = 0;
    _Float16* wpack = (_Float16*)(ws + off); off += (size_t)4 * CDIM * DDIM * sizeof(_Float16);
    off = (off + 255) & ~(size_t)255;
    float* q   = (float*)(ws + off); off += (size_t)N * DDIM * sizeof(float);
    float* k   = (float*)(ws + off); off += (size_t)N * DDIM * sizeof(float);
    float* v   = (float*)(ws + off); off += (size_t)N * DDIM * sizeof(float);
    float* xr  = (float*)(ws + off); off += (size_t)N * DDIM * sizeof(float);
    float* agg = (float*)(ws + off); off += (size_t)N * DDIM * sizeof(float);
    float* logits = (float*)(ws + off); off += (size_t)E * HEADS * sizeof(float);
    float*    nodesum = (float*)   (ws + off); off += (size_t)N * HEADS * sizeof(float);
    unsigned* nodemax = (unsigned*)(ws + off); off += (size_t)N * HEADS * sizeof(unsigned);
    (void)ws_size;

    float* coopO = (float*)d_out;
    float* anomO = coopO + N;
    float* hO    = coopO + 2 * (size_t)N;

    // 1) pack weights into WMMA-B layout
    pack_w_kernel<<<(4 * CDIM * DDIM + 255) / 256, 256, 0, stream>>>(
        Wq, Wk, Wv, Wskip, wpack);

    // 2) embedding + q/k/v/x_r projections (WMMA)
    qkv_wmma_kernel<<<(N + 15) / 16, 256, 0, stream>>>(
        x, embed, wpack, bq, bk, bv, bskip, q, k, v, xr, N);

    // 3) init aggregation + softmax stats
    {
        int n256 = N * DDIM, n4 = N * HEADS;
        init_kernel<<<(n256 + 255) / 256, 256, 0, stream>>>(agg, nodesum, nodemax, n256, n4);
    }

    // 4) logits + segment max
    edge_logits_kernel<<<((size_t)E * HEADS + 255) / 256, 256, 0, stream>>>(
        ei, q, k, logits, nodemax, E);

    // 5) exp + segment sum
    edge_exp_kernel<<<((size_t)E * HEADS + 255) / 256, 256, 0, stream>>>(
        ei, logits, nodemax, nodesum, E);

    // 6) weighted scatter-aggregate
    edge_agg_kernel<<<(E + 7) / 8, 256, 0, stream>>>(
        ei, logits, nodesum, v, agg, E);

    // 7) beta gate + ReLU + heads
    finalize_kernel<<<(N + 7) / 8, 256, 0, stream>>>(
        agg, xr, Wbeta, Wcoop, bcoop, Wanom, banom, coopO, anomO, hO, N);
}
